// GAT_77129022701601
// MI455X (gfx1250) — matrix operand order repaired and checked by reading, not compiled, not run on hardware
//
#include <hip/hip_runtime.h>

#define IN_DIM 512
#define F1     256     // H1 * HID
#define H1C    64      // HID
#define H1N    4       // heads layer 1
#define NCLS   40

typedef __attribute__((ext_vector_type(16))) __bf16       v16bf;
typedef __attribute__((ext_vector_type(8)))  float        v8f;
typedef __attribute__((ext_vector_type(4)))  unsigned int v4u;
typedef __attribute__((ext_vector_type(8)))  unsigned int v8u;

// ---------- helpers ----------
__device__ __forceinline__ unsigned short f2bf(float f) {
  unsigned int u = __float_as_uint(f);
  unsigned int r = u + 0x7FFFu + ((u >> 16) & 1u);      // round-to-nearest-even
  if ((u & 0x7F800000u) == 0x7F800000u) r = u;          // inf/nan: truncate
  return (unsigned short)(r >> 16);
}
// order-preserving float -> uint key (for atomic max); key(x) monotone in x
__device__ __forceinline__ unsigned int fkey(float f) {
  unsigned int u = __float_as_uint(f);
  return (u & 0x80000000u) ? ~u : (u | 0x80000000u);
}
__device__ __forceinline__ float dkey(unsigned int k) {
  unsigned int u = (k & 0x80000000u) ? (k ^ 0x80000000u) : ~k;
  return __uint_as_float(u);
}

// ---------- prep kernels ----------
__global__ void k_cvt_bf16(const float* __restrict__ in, unsigned short* __restrict__ out, int n) {
  int t = blockIdx.x * blockDim.x + threadIdx.x;
  if (t < n) out[t] = f2bf(in[t]);
}

__global__ void k_zero4(float* __restrict__ p, int n4) {
  int t = blockIdx.x * blockDim.x + threadIdx.x;
  if (t < n4) ((float4*)p)[t] = make_float4(0.f, 0.f, 0.f, 0.f);
}

// Pack W [K x Ncols row-major, f32] into per-tile WMMA B-fragment order (bf16).
// packed[((kt*Ntiles+nt)*32 + lane)*16 + elem]; h=lane>>4: elem<8 -> k=8h+elem, elem>=8 -> k=16+8h+(elem-8); n=lane&15
__global__ void k_pack_B(const float* __restrict__ W, unsigned short* __restrict__ out,
                         int K, int Ncols, int Ntiles) {
  int t = blockIdx.x * blockDim.x + threadIdx.x;
  int total = (K >> 5) * Ntiles * 512;
  if (t >= total) return;
  int elem = t & 15;
  int lane = (t >> 4) & 31;
  int tile = t >> 9;
  int kt = tile / Ntiles, nt = tile % Ntiles;
  int k = kt * 32 + ((lane >> 4) << 3) + ((elem < 8) ? elem : 8 + elem);
  int ncol = nt * 16 + (lane & 15);
  float v = (ncol < Ncols) ? W[(size_t)k * Ncols + ncol] : 0.f;
  out[t] = f2bf(v);
}

// ---------- WMMA GEMM: C[M x ldC] = A[M x K](bf16, row-major) * Bpacked ----------
template <int KTILES>
__global__ void k_gemm_bf16(const unsigned short* __restrict__ A,
                            const unsigned short* __restrict__ Bp,
                            float* __restrict__ C,
                            int K, int Ntiles, int Nstore, int ldC) {
  const int mTile = blockIdx.x;
  const int nTile = blockIdx.y;
  const int lane  = threadIdx.x;              // blockDim.x == 32, one wave
  const int row   = mTile * 16 + (lane & 15);
  const int kHalf = ((lane >> 4) & 1) * 8;
  const unsigned short* arow = A + (size_t)row * K;

  v8f acc = {};
#pragma unroll
  for (int kt = 0; kt < KTILES; ++kt) {
    int k0 = kt * 32 + kHalf;
    v4u alo = *(const v4u*)(arow + k0);        // K: k0..k0+7
    v4u ahi = *(const v4u*)(arow + k0 + 16);   // K: k0+16..k0+23
    v8u av = {alo.x, alo.y, alo.z, alo.w, ahi.x, ahi.y, ahi.z, ahi.w};
    v8u bv = *(const v8u*)(Bp + (((size_t)(kt * Ntiles + nTile) * 32 + lane) << 4));
    v16bf a = __builtin_bit_cast(v16bf, av);
    v16bf b = __builtin_bit_cast(v16bf, bv);
    acc = __builtin_amdgcn_wmma_f32_16x16x32_bf16(false, a, false, b,
                                                  (short)0, acc, false, false);
  }
  // C/D layout: lane<16 -> (M=r, N=lane); lane>=16 -> (M=8+r, N=lane-16)
  int col   = nTile * 16 + (lane & 15);
  int rbase = (lane < 16) ? 0 : 8;
  if (col < Nstore) {
#pragma unroll
    for (int r = 0; r < 8; ++r)
      C[(size_t)(mTile * 16 + rbase + r) * ldC + col] = acc[r];
  }
}

// ---------- attention scores: a_s[n,h], a_d[n,h] ----------
__global__ void k_att(const float* __restrict__ H, const float* __restrict__ att_src,
                      const float* __restrict__ att_dst, float* __restrict__ as_,
                      float* __restrict__ ad_, int n, int heads, int ch) {
  int t = blockIdx.x * blockDim.x + threadIdx.x;
  if (t >= n * heads) return;
  int node = t / heads, h = t % heads;
  const float* row = H + (size_t)node * heads * ch + h * ch;
  const float* vs = att_src + h * ch;
  const float* vd = att_dst + h * ch;
  float ss = 0.f, sd = 0.f;
  for (int c = 0; c < ch; ++c) { float v = row[c]; ss += v * vs[c]; sd += v * vd[c]; }
  as_[t] = ss; ad_[t] = sd;
}

// ---------- edge pass A: e = leaky_relu(as[src]+ad[dst]); segment max ----------
__global__ void k_edge_a(const int* __restrict__ ei, const float* __restrict__ as_,
                         const float* __restrict__ ad_, float* __restrict__ e,
                         unsigned int* __restrict__ mkey, int E_, int n, int heads) {
  int t = blockIdx.x * blockDim.x + threadIdx.x;
  int Etot = E_ + n;
  if (t >= Etot * heads) return;
  int edge = t / heads, h = t % heads;
  int src, dst;
  if (edge < E_) { src = ei[edge]; dst = ei[E_ + edge]; }
  else           { src = dst = edge - E_; }                 // self loop
  float v = as_[src * heads + h] + ad_[dst * heads + h];
  v = (v > 0.f) ? v : 0.2f * v;                             // leaky relu
  e[t] = v;
  __hip_atomic_fetch_max(&mkey[dst * heads + h], fkey(v),
                         __ATOMIC_RELAXED, __HIP_MEMORY_SCOPE_AGENT);
}

// ---------- edge pass B: ex = exp(e - m[dst]); segment sum ----------
__global__ void k_edge_b(const int* __restrict__ ei, float* __restrict__ e,
                         const unsigned int* __restrict__ mkey, float* __restrict__ s,
                         int E_, int n, int heads) {
  int t = blockIdx.x * blockDim.x + threadIdx.x;
  int Etot = E_ + n;
  if (t >= Etot * heads) return;
  int edge = t / heads, h = t % heads;
  int dst = (edge < E_) ? ei[E_ + edge] : (edge - E_);
  float m = dkey(mkey[dst * heads + h]);
  float ex = expf(e[t] - m);
  e[t] = ex;
  __hip_atomic_fetch_add(&s[dst * heads + h], ex,
                         __ATOMIC_RELAXED, __HIP_MEMORY_SCOPE_AGENT);
}

// ---------- edge pass C: out[dst] += alpha * H[src]; one wave per edge ----------
__global__ void k_edge_c(const int* __restrict__ ei, const float* __restrict__ ex,
                         const float* __restrict__ s, const float* __restrict__ H,
                         float* __restrict__ out, int E_, int n, int heads, int ch) {
  int wid  = (blockIdx.x * blockDim.x + threadIdx.x) >> 5;
  int lane = threadIdx.x & 31;
  int Etot = E_ + n;
  if (wid >= Etot) return;
  int src, dst;
  if (wid < E_) { src = ei[wid]; dst = ei[E_ + wid]; }
  else          { src = dst = wid - E_; }
  int F = heads * ch;
  const float* hrow = H + (size_t)src * F;
  float* orow = out + (size_t)dst * F;
  for (int c = lane; c < F; c += 32) {
    int h = c / ch;
    float alpha = ex[(size_t)wid * heads + h] / (s[dst * heads + h] + 1e-16f);
    __hip_atomic_fetch_add(&orow[c], alpha * hrow[c],
                           __ATOMIC_RELAXED, __HIP_MEMORY_SCOPE_AGENT);
  }
}

// ---------- bias + ELU -> bf16 (input to GEMM2) ----------
__global__ void k_bias_elu(const float* __restrict__ in, const float* __restrict__ b,
                           unsigned short* __restrict__ outb, int n, int F) {
  int t = blockIdx.x * blockDim.x + threadIdx.x;
  if (t >= n * F) return;
  float v = in[t] + b[t % F];
  v = (v > 0.f) ? v : (expf(v) - 1.f);
  outb[t] = f2bf(v);
}

// ---------- final: bias + log_softmax over NCLS ----------
__global__ void k_lsm(const float* __restrict__ in, const float* __restrict__ b,
                      float* __restrict__ y, int n) {
  int node = blockIdx.x * blockDim.x + threadIdx.x;
  if (node >= n) return;
  float v[NCLS];
  float mx = -3.4e38f;
  for (int c = 0; c < NCLS; ++c) { v[c] = in[(size_t)node * NCLS + c] + b[c]; mx = fmaxf(mx, v[c]); }
  float sum = 0.f;
  for (int c = 0; c < NCLS; ++c) sum += expf(v[c] - mx);
  float ls = mx + logf(sum);
  for (int c = 0; c < NCLS; ++c) y[(size_t)node * NCLS + c] = v[c] - ls;
}

extern "C" void kernel_launch(void* const* d_in, const int* in_sizes, int n_in,
                              void* d_out, int out_size, void* d_ws, size_t ws_size,
                              hipStream_t stream) {
  const float* x        = (const float*)d_in[0];
  const int*   ei       = (const int*)  d_in[1];
  const float* W1       = (const float*)d_in[2];
  const float* att_src1 = (const float*)d_in[3];
  const float* att_dst1 = (const float*)d_in[4];
  const float* b1       = (const float*)d_in[5];
  const float* W2       = (const float*)d_in[6];
  const float* att_src2 = (const float*)d_in[7];
  const float* att_dst2 = (const float*)d_in[8];
  const float* b2       = (const float*)d_in[9];
  float* y = (float*)d_out;

  const int n    = in_sizes[0] / IN_DIM;   // 50000 (multiple of 16)
  const int E_   = in_sizes[1] / 2;        // 800000
  const int Etot = E_ + n;

  // workspace layout (256B aligned regions)
  char* ws = (char*)d_ws;
  size_t off = 0;
  auto take = [&](size_t bytes) { size_t o = off; off += (bytes + 255) & ~(size_t)255; return o; };
  size_t o_xb  = take((size_t)n * IN_DIM * 2);   // bf16 x; reused for bf16 elu(h1)
  size_t o_w1p = take((size_t)16 * 16 * 512 * 2);
  size_t o_w2p = take((size_t)8 * 3 * 512 * 2);
  size_t o_h   = take((size_t)n * F1 * 4);       // h1 f32; reused for h2 [n x 40]
  size_t o_as  = take((size_t)n * H1N * 4);
  size_t o_ad  = take((size_t)n * H1N * 4);
  size_t o_mk  = take((size_t)n * H1N * 4);
  size_t o_s   = take((size_t)n * H1N * 4);
  size_t o_e   = take((size_t)Etot * H1N * 4);
  size_t o_out = take((size_t)n * F1 * 4);       // out1 f32; reused for out2 [n x 40]

  unsigned short* xb  = (unsigned short*)(ws + o_xb);
  unsigned short* w1p = (unsigned short*)(ws + o_w1p);
  unsigned short* w2p = (unsigned short*)(ws + o_w2p);
  float* h   = (float*)(ws + o_h);
  float* as_ = (float*)(ws + o_as);
  float* ad_ = (float*)(ws + o_ad);
  unsigned int* mk = (unsigned int*)(ws + o_mk);
  float* s   = (float*)(ws + o_s);
  float* e   = (float*)(ws + o_e);
  float* out = (float*)(ws + o_out);

  const int B = 256;
  auto g = [](int work, int blk) { return (work + blk - 1) / blk; };

  // ---- prep ----
  k_cvt_bf16<<<g(n * IN_DIM, B), B, 0, stream>>>(x, xb, n * IN_DIM);
  k_pack_B<<<g(16 * 16 * 512, B), B, 0, stream>>>(W1, w1p, IN_DIM, F1, 16);
  k_pack_B<<<g(8 * 3 * 512, B), B, 0, stream>>>(W2, w2p, F1, NCLS, 3);

  // ---- layer 1 ----
  k_zero4<<<g(n * H1N / 4, B), B, 0, stream>>>((float*)mk, n * H1N / 4);
  k_zero4<<<g(n * H1N / 4, B), B, 0, stream>>>(s, n * H1N / 4);
  k_zero4<<<g(n * F1 / 4, B), B, 0, stream>>>(out, n * F1 / 4);

  dim3 g1(n / 16, 16);
  k_gemm_bf16<16><<<g1, 32, 0, stream>>>(xb, w1p, h, IN_DIM, 16, F1, F1);

  k_att<<<g(n * H1N, B), B, 0, stream>>>(h, att_src1, att_dst1, as_, ad_, n, H1N, H1C);
  k_edge_a<<<g(Etot * H1N, B), B, 0, stream>>>(ei, as_, ad_, e, mk, E_, n, H1N);
  k_edge_b<<<g(Etot * H1N, B), B, 0, stream>>>(ei, e, mk, s, E_, n, H1N);
  k_edge_c<<<g(Etot * 32, B), B, 0, stream>>>(ei, e, s, h, out, E_, n, H1N, H1C);

  unsigned short* hb = xb;  // reuse region: bf16 elu(h1) [n x 256]
  k_bias_elu<<<g(n * F1, B), B, 0, stream>>>(out, b1, hb, n, F1);

  // ---- layer 2 ----
  k_zero4<<<g(n / 4, B), B, 0, stream>>>((float*)mk, n / 4);
  k_zero4<<<g(n / 4, B), B, 0, stream>>>(s, n / 4);
  k_zero4<<<g(n * NCLS / 4, B), B, 0, stream>>>(out, n * NCLS / 4);

  dim3 g2(n / 16, 3);
  k_gemm_bf16<8><<<g2, 32, 0, stream>>>(hb, w2p, h, F1, 3, NCLS, NCLS);

  k_att<<<g(n, B), B, 0, stream>>>(h, att_src2, att_dst2, as_, ad_, n, 1, NCLS);
  k_edge_a<<<g(Etot, B), B, 0, stream>>>(ei, as_, ad_, e, mk, E_, n, 1);
  k_edge_b<<<g(Etot, B), B, 0, stream>>>(ei, e, mk, s, E_, n, 1);
  k_edge_c<<<g(Etot * 32, B), B, 0, stream>>>(ei, e, s, h, out, E_, n, 1, NCLS);

  k_lsm<<<g(n, B), B, 0, stream>>>(out, b2, y, n);
}
